// CRVAE_15075335209044
// MI455X (gfx1250) — matrix operand order repaired and checked by reading, not compiled, not run on hardware
//
#include <hip/hip_runtime.h>
#include <hip/hip_bf16.h>

// ---------------------------------------------------------------------------
// CR-VAE on MI455X (gfx1250): bf16 WMMA GRU, fp32 accumulate.
// B=32, TAU=128, D=64, H=256, Z=128.
// Decoder: one 512-thread block per head (16 waves = 16 column-groups).
// Weights stream from the 192MB L2 each step; an opaque SGPR offset (not a
// laundered pointer, which degraded loads to flat_*) defeats LICM hoisting
// while keeping the loads on the global_load_b128 / LOADcnt path.
// ---------------------------------------------------------------------------

typedef __bf16 v16bf __attribute__((ext_vector_type(16)));
typedef float  v8f   __attribute__((ext_vector_type(8)));

#define BQ 32
#define TAU 128
#define DQ 64
#define HQ 256
#define ZQ 128
#define KT_TOT 10      // 10 K-tiles of 32 (2 for x, 8 for h)
#define NT_TOT 48      // 48 N-tiles of 16 (3 gates x 16 groups)
#define HS_STRIDE 264  // padded bf16 stride: 528B rows, 16B aligned, bank-spread

// ---- bf16 helpers (RNE) ----------------------------------------------------
__device__ inline unsigned short bf16_of(float f) {
  unsigned int u = __float_as_uint(f);
  u += 0x7FFFu + ((u >> 16) & 1u);
  return (unsigned short)(u >> 16);
}
__device__ inline float f_of_bf16(unsigned short h) {
  return __uint_as_float(((unsigned int)h) << 16);
}
__device__ inline unsigned int pk_bf16(float lo, float hi) {
  return (unsigned int)bf16_of(lo) | ((unsigned int)bf16_of(hi) << 16);
}
__device__ inline float sigm(float x) { return 1.0f / (1.0f + __expf(-x)); }

union FragU { uint4 u[2]; v16bf v; };

// Load a pre-packed fragment block (64 uint4 = 1KB, lane-major 32B/lane).
__device__ inline v16bf load_frag_g(const uint4* __restrict__ base, int lane) {
  FragU x;
  x.u[0] = base[lane * 2 + 0];
  x.u[1] = base[lane * 2 + 1];
  return x.v;
}

// A-fragment of the hidden state from LDS (bf16 row-major, padded stride).
// 16-bit A 16x32 layout: lanes 0-15: V0-3 K0..7, V4-7 K16..23;
//                        lanes16-31: V0-3 K8..15, V4-7 K24..31.
__device__ inline v16bf load_hfrag(const unsigned short* hS, int mt, int kt, int lane) {
  int lr = lane & 15, hl = lane >> 4;
  const char* rowp = (const char*)(hS + (mt * 16 + lr) * HS_STRIDE);
  int off = (kt * 32 + hl * 8) * 2;
  FragU x;
  x.u[0] = *(const uint4*)(rowp + off);
  x.u[1] = *(const uint4*)(rowp + off + 32);
  return x.v;
}

// ---------------------------------------------------------------------------
// Pack x (fp32 [B,TAU,D]) into A-fragment blocks: frag id = (t*2+mt)*2+kx.
// ---------------------------------------------------------------------------
__global__ void pack_x_kernel(const float* __restrict__ xe,
                              const float* __restrict__ xr,
                              unsigned int* __restrict__ dst) {
  int bid = blockIdx.x;            // 0..1023
  int which = bid >> 9;            // 0: x_early, 1: x_recent
  int f = bid & 511;               // (t*2+mt)*2+kx
  int kx = f & 1, mtt = (f >> 1) & 1, t = f >> 2;
  const float* src = which ? xr : xe;
  unsigned int* out = dst + (size_t)which * (512 * 256) + (size_t)f * 256;
  int lane = threadIdx.x & 31;
  int lr = lane & 15, hl = lane >> 4;
  int b = mtt * 16 + lr;
#pragma unroll
  for (int j = 0; j < 8; ++j) {
    int K = ((j < 4) ? (2 * j) : (16 + 2 * (j - 4))) + hl * 8 + kx * 32;
    float v0 = src[((size_t)b * TAU + t) * DQ + K];
    float v1 = src[((size_t)b * TAU + t) * DQ + K + 1];
    out[lane * 8 + j] = pk_bf16(v0, v1);
  }
}

// ---------------------------------------------------------------------------
// Pack concatenated weights Wcat = [Wih_eff ; Whh] (K=320, N=768) into
// B-fragment blocks, bf16. heads 0..63 = decoder (mask folded), 64 = encoder.
// 16-bit B 32x16 layout: lanes 0-15: V0-3 K0..7, V4-7 K16..23; lanes 16-31: V0-3 K8..15, V4-7 K24..31.
// ---------------------------------------------------------------------------
__global__ void pack_w_kernel(const float* __restrict__ Aadj,
                              const float* __restrict__ encWih,
                              const float* __restrict__ encWhh,
                              const float* __restrict__ hWih,
                              const float* __restrict__ hWhh,
                              unsigned int* __restrict__ dst) {
  int bid = blockIdx.x;            // head*480 + kt*48 + nt, head 0..64
  int head = bid / 480;
  int r = bid % 480;
  int kt = r / 48, nt = r % 48;
  int lane = threadIdx.x & 31;
  int lr = lane & 15, hl = lane >> 4;
  int n = nt * 16 + lr;
  unsigned int* out = dst + (size_t)bid * 256;
#pragma unroll
  for (int j = 0; j < 8; ++j) {
    int K0 = kt * 32 + hl * 8 + ((j < 4) ? (2 * j) : (16 + 2 * (j - 4)));
    float v[2];
#pragma unroll
    for (int s = 0; s < 2; ++s) {
      int K = K0 + s;
      float w;
      if (K < DQ) {
        if (head < 64) {
          float m = (fabsf(Aadj[head * DQ + K]) > 1e-6f) ? 1.0f : 0.0f;
          w = hWih[((size_t)head * 768 + n) * DQ + K] * m;
        } else {
          w = encWih[(size_t)n * DQ + K];
        }
      } else {
        int Kh = K - DQ;
        if (head < 64) w = hWhh[((size_t)head * 768 + n) * HQ + Kh];
        else           w = encWhh[(size_t)n * HQ + Kh];
      }
      v[s] = w;
    }
    out[lane * 8 + j] = pk_bf16(v[0], v[1]);
  }
}

// ---------------------------------------------------------------------------
// GRU scan kernel (shared by encoder and decoder).
// Block = 512 threads = 16 waves; wave w owns column-group g=w (16 H-cols),
// computing r/z/n gate tiles for both M-tiles across K=320 (10 K-tiles).
// Decoder: grid = 64 heads. Encoder: grid = 1 (head slot 64 in packed W).
// ---------------------------------------------------------------------------
__global__ void __launch_bounds__(512)
gru_wmma_kernel(const uint4* __restrict__ xfrag,   // frag blocks for this input
                const uint4* __restrict__ wfrag,   // all packed weights
                const float* __restrict__ bihAll,
                const float* __restrict__ bhhAll,
                const float* __restrict__ h0,      // [32,256] fp32 or null
                const float* __restrict__ woutAll, // [64,256] (decoder)
                const float* __restrict__ boutAll, // [64]     (decoder)
                float* __restrict__ recon,         // [B,TAU,D] (decoder)
                float* __restrict__ hout,          // [32,256]  (encoder)
                int wHeadBase, int biasStride, int isDec) {
  __shared__ unsigned short hS[32 * HS_STRIDE];  // bf16 hidden, 32 rows
  __shared__ float bI[768], bH[768];
  __shared__ float woutS[HQ];

  const int tid = threadIdx.x;
  const int wave = tid >> 5, lane = tid & 31;
  const int lr = lane & 15, hl = lane >> 4;
  const int head = isDec ? blockIdx.x : 0;
  const int g = wave;  // column-group 0..15

  const float* bih = bihAll + (size_t)head * biasStride;
  const float* bhh = bhhAll + (size_t)head * biasStride;
  for (int i = tid; i < 768; i += 512) { bI[i] = bih[i]; bH[i] = bhh[i]; }
  if (isDec) {
    for (int i = tid; i < HQ; i += 512) woutS[i] = woutAll[(size_t)head * HQ + i];
  }
  for (int i = tid; i < 32 * HQ; i += 512) {
    int r = i >> 8, c = i & 255;
    float v = h0 ? h0[(size_t)r * HQ + c] : 0.0f;
    hS[r * HS_STRIDE + c] = bf16_of(v);
  }
  __syncthreads();

  const uint4* wbase = wfrag + (size_t)(wHeadBase + head * (KT_TOT * NT_TOT)) * 64;

  for (int t = 0; t < TAU; ++t) {
    // Opaque SGPR zero, refreshed every step: defeats LICM (so the 30
    // B-fragments stream from L2 each iteration instead of being hoisted
    // into spilled registers) while preserving the pointer's global
    // address-space provenance (round 2's pointer laundering degraded the
    // weight stream to flat_load_b128, coupling it to DScnt).
    int opq = 0;
    asm volatile("" : "+s"(opq));
    const uint4* wt = wbase + opq;

    v8f accR[2]  = {{0,0,0,0,0,0,0,0}, {0,0,0,0,0,0,0,0}};
    v8f accZ[2]  = {{0,0,0,0,0,0,0,0}, {0,0,0,0,0,0,0,0}};
    v8f accNx[2] = {{0,0,0,0,0,0,0,0}, {0,0,0,0,0,0,0,0}};
    v8f accNh[2] = {{0,0,0,0,0,0,0,0}, {0,0,0,0,0,0,0,0}};

    // x-part: K-tiles 0..1 feed all three gates (n gets its own acc).
#pragma unroll
    for (int kx = 0; kx < 2; ++kx) {
      v16bf a0 = load_frag_g(xfrag + (size_t)((t * 2 + 0) * 2 + kx) * 64, lane);
      v16bf a1 = load_frag_g(xfrag + (size_t)((t * 2 + 1) * 2 + kx) * 64, lane);
      v16bf bR = load_frag_g(wt + (size_t)(kx * NT_TOT + g) * 64, lane);
      v16bf bZ = load_frag_g(wt + (size_t)(kx * NT_TOT + 16 + g) * 64, lane);
      v16bf bN = load_frag_g(wt + (size_t)(kx * NT_TOT + 32 + g) * 64, lane);
      accR[0]  = __builtin_amdgcn_wmma_f32_16x16x32_bf16(false, a0, false, bR, (short)0, accR[0],  false, false);
      accR[1]  = __builtin_amdgcn_wmma_f32_16x16x32_bf16(false, a1, false, bR, (short)0, accR[1],  false, false);
      accZ[0]  = __builtin_amdgcn_wmma_f32_16x16x32_bf16(false, a0, false, bZ, (short)0, accZ[0],  false, false);
      accZ[1]  = __builtin_amdgcn_wmma_f32_16x16x32_bf16(false, a1, false, bZ, (short)0, accZ[1],  false, false);
      accNx[0] = __builtin_amdgcn_wmma_f32_16x16x32_bf16(false, a0, false, bN, (short)0, accNx[0], false, false);
      accNx[1] = __builtin_amdgcn_wmma_f32_16x16x32_bf16(false, a1, false, bN, (short)0, accNx[1], false, false);
    }
    // h-part: K-tiles 2..9; n-gate recurrent part accumulated separately.
#pragma unroll
    for (int kh = 0; kh < 8; ++kh) {
      int kt = 2 + kh;
      v16bf a0 = load_hfrag(hS, 0, kh, lane);
      v16bf a1 = load_hfrag(hS, 1, kh, lane);
      v16bf bR = load_frag_g(wt + (size_t)(kt * NT_TOT + g) * 64, lane);
      v16bf bZ = load_frag_g(wt + (size_t)(kt * NT_TOT + 16 + g) * 64, lane);
      v16bf bN = load_frag_g(wt + (size_t)(kt * NT_TOT + 32 + g) * 64, lane);
      accR[0]  = __builtin_amdgcn_wmma_f32_16x16x32_bf16(false, a0, false, bR, (short)0, accR[0],  false, false);
      accR[1]  = __builtin_amdgcn_wmma_f32_16x16x32_bf16(false, a1, false, bR, (short)0, accR[1],  false, false);
      accZ[0]  = __builtin_amdgcn_wmma_f32_16x16x32_bf16(false, a0, false, bZ, (short)0, accZ[0],  false, false);
      accZ[1]  = __builtin_amdgcn_wmma_f32_16x16x32_bf16(false, a1, false, bZ, (short)0, accZ[1],  false, false);
      accNh[0] = __builtin_amdgcn_wmma_f32_16x16x32_bf16(false, a0, false, bN, (short)0, accNh[0], false, false);
      accNh[1] = __builtin_amdgcn_wmma_f32_16x16x32_bf16(false, a1, false, bN, (short)0, accNh[1], false, false);
    }

    __syncthreads();  // all hS reads of this step complete

    // GRU nonlinearity; C layout: VGPR i, lanes 0-15 -> row i, lanes 16-31 -> row i+8.
    const int hc = g * 16 + lr;  // hidden column owned by this lane
#pragma unroll
    for (int mt = 0; mt < 2; ++mt) {
#pragma unroll
      for (int i = 0; i < 8; ++i) {
        int row = mt * 16 + i + hl * 8;  // batch row (0..31)
        float rg = sigm(accR[mt][i] + bI[hc] + bH[hc]);
        float zg = sigm(accZ[mt][i] + bI[256 + hc] + bH[256 + hc]);
        float ng = tanhf(accNx[mt][i] + bI[512 + hc] + rg * (accNh[mt][i] + bH[512 + hc]));
        float hprev = f_of_bf16(hS[row * HS_STRIDE + hc]);
        float hn = (1.0f - zg) * ng + zg * hprev;
        hS[row * HS_STRIDE + hc] = bf16_of(hn);
      }
    }
    __syncthreads();  // hS fully updated

    if (isDec && tid < 32) {
      // Deterministic per-row reduction: recon[b,t,head] = h . Wout[head] + bout
      const unsigned int* rowp = (const unsigned int*)(hS + tid * HS_STRIDE);
      float s = 0.0f;
#pragma unroll 8
      for (int c2 = 0; c2 < HQ / 2; ++c2) {
        unsigned int pv = rowp[c2];
        s += __uint_as_float(pv << 16) * woutS[2 * c2];
        s += __uint_as_float(pv & 0xFFFF0000u) * woutS[2 * c2 + 1];
      }
      recon[((size_t)tid * TAU + t) * DQ + head] = s + boutAll[head];
    }
    // next step's hS writes are fenced by the post-K-loop __syncthreads()
  }

  if (!isDec) {
    for (int i = tid; i < 32 * HQ; i += 512) {
      int r = i >> 8, c = i & 255;
      hout[(size_t)r * HQ + c] = f_of_bf16(hS[r * HS_STRIDE + c]);
    }
  }
}

// ---------------------------------------------------------------------------
// Latent head: mu, logsig, z  (4096 outputs, trivially small)
// ---------------------------------------------------------------------------
__global__ void latent1_kernel(const float* __restrict__ hEnc,
                               const float* __restrict__ Wmu, const float* __restrict__ bmu,
                               const float* __restrict__ Wls, const float* __restrict__ bls,
                               const float* __restrict__ eps,
                               float* __restrict__ outMu, float* __restrict__ outLs,
                               float* __restrict__ zBuf) {
  int idx = blockIdx.x * blockDim.x + threadIdx.x;
  if (idx >= BQ * ZQ) return;
  int b = idx >> 7, j = idx & 127;
  float sm = 0.0f, sl = 0.0f;
#pragma unroll 4
  for (int k = 0; k < HQ; ++k) {
    float h = hEnc[b * HQ + k];
    sm += h * Wmu[j * HQ + k];
    sl += h * Wls[j * HQ + k];
  }
  sm += bmu[j]; sl += bls[j];
  outMu[idx] = sm;
  outLs[idx] = sl;
  zBuf[idx] = sm + eps[idx] * __expf(sl);
}

__global__ void latent2_kernel(const float* __restrict__ zBuf,
                               const float* __restrict__ Wre, const float* __restrict__ bre,
                               float* __restrict__ h0) {
  int idx = blockIdx.x * blockDim.x + threadIdx.x;
  if (idx >= BQ * HQ) return;
  int b = idx >> 8, c = idx & 255;
  float s = bre[c];
#pragma unroll 4
  for (int k = 0; k < ZQ; ++k) s += zBuf[b * ZQ + k] * Wre[c * ZQ + k];
  h0[idx] = tanhf(s);
}

// ---------------------------------------------------------------------------
extern "C" void kernel_launch(void* const* d_in, const int* in_sizes, int n_in,
                              void* d_out, int out_size, void* d_ws, size_t ws_size,
                              hipStream_t stream) {
  const float* x_early  = (const float*)d_in[0];
  const float* x_recent = (const float*)d_in[1];
  const float* eps      = (const float*)d_in[2];
  const float* Aadj     = (const float*)d_in[3];
  const float* enc_Wih  = (const float*)d_in[4];
  const float* enc_Whh  = (const float*)d_in[5];
  const float* enc_bih  = (const float*)d_in[6];
  const float* enc_bhh  = (const float*)d_in[7];
  const float* W_mu     = (const float*)d_in[8];
  const float* b_mu     = (const float*)d_in[9];
  const float* W_ls     = (const float*)d_in[10];
  const float* b_ls     = (const float*)d_in[11];
  const float* W_re     = (const float*)d_in[12];
  const float* b_re     = (const float*)d_in[13];
  const float* h_Wih    = (const float*)d_in[14];
  const float* h_Whh    = (const float*)d_in[15];
  const float* h_bih    = (const float*)d_in[16];
  const float* h_bhh    = (const float*)d_in[17];
  const float* h_Wout   = (const float*)d_in[18];
  const float* h_bout   = (const float*)d_in[19];
  float* out = (float*)d_out;
  char* ws = (char*)d_ws;

  // Workspace layout (bytes):
  //  [0, 1MB)                 : x fragments (x_early then x_recent), bf16
  //  [1MB, 1MB+31200KB)       : packed Wcat fragments for 64 heads + encoder
  //  then: hEnc (32KB), zBuf (16KB), h0 (32KB)
  unsigned int* xfrag = (unsigned int*)ws;
  size_t off = (size_t)1024 * 1024;
  unsigned int* wfrag = (unsigned int*)(ws + off);
  off += (size_t)65 * 480 * 1024;
  float* hEnc = (float*)(ws + off); off += (size_t)BQ * HQ * 4;
  float* zBuf = (float*)(ws + off); off += (size_t)BQ * ZQ * 4;
  float* h0   = (float*)(ws + off);

  // 1) one-time packs (x -> A-fragments, weights -> B-fragments, mask folded)
  pack_x_kernel<<<1024, 32, 0, stream>>>(x_early, x_recent, xfrag);
  pack_w_kernel<<<65 * 480, 32, 0, stream>>>(Aadj, enc_Wih, enc_Whh, h_Wih, h_Whh, wfrag);

  // 2) encoder GRU scan (head slot 64 in packed weights), h0 = 0
  gru_wmma_kernel<<<1, 512, 0, stream>>>(
      (const uint4*)xfrag, (const uint4*)wfrag,
      enc_bih, enc_bhh, nullptr, nullptr, nullptr,
      nullptr, hEnc, /*wHeadBase=*/64 * 480, /*biasStride=*/0, /*isDec=*/0);

  // 3) latent: mu, logsig -> d_out; z -> ws; h0 = tanh(z W_re^T + b_re)
  latent1_kernel<<<16, 256, 0, stream>>>(hEnc, W_mu, b_mu, W_ls, b_ls, eps,
                                         out + 262144, out + 262144 + 4096, zBuf);
  latent2_kernel<<<32, 256, 0, stream>>>(zBuf, W_re, b_re, h0);

  // 4) 64 decoder heads (one block each), recon -> d_out[0:262144]
  gru_wmma_kernel<<<64, 512, 0, stream>>>(
      (const uint4*)(xfrag + 512 * 256), (const uint4*)wfrag,
      h_bih, h_bhh, h0, h_Wout, h_bout,
      out, nullptr, /*wHeadBase=*/0, /*biasStride=*/768, /*isDec=*/1);
}
